// AttentionConv1d_26336739459333
// MI455X (gfx1250) — matrix operand order repaired and checked by reading, not compiled, not run on hardware
//
#include <hip/hip_runtime.h>
#include <hip/hip_bf16.h>

// ---------------------------------------------------------------------------
// AttentionConv1d on MI455X (gfx1250):  out[b,o,t] = sum_i W[o,i] * G[i,t]
//   G[i,t] = feature_pad[b, i/3, t + i%3 - 1] * att[b, i>>9, t]
// GEMM via v_wmma_f32_16x16x32_bf16 (bf16 hi/lo split, ~fp32 accuracy).
// A-tiles streamed to LDS with global_load_async_to_lds_b128 (ASYNCcnt).
// K-loop unrolled by 3 so im2col (c,k) offsets are loop-invariant registers.
// ---------------------------------------------------------------------------

typedef __bf16 bf16;
typedef __attribute__((ext_vector_type(16))) __bf16 v16bf;
typedef __attribute__((ext_vector_type(8)))  __bf16 v8bf;
typedef __attribute__((ext_vector_type(8)))  float  v8f;

#define B_  8
#define C_  512
#define T_  4096
#define O_  512
#define KK_ 1536          // 3*C
#define EPS 1e-8f

#define BM 256            // output tile: O
#define BN 64             // output tile: T
#define BK 32             // K step (one bf16 wmma K)
#define LDA 40            // LDS row stride in bf16 (80B: 16B aligned, conflict-friendly)

// ---------------------------------------------------------------------------
// Kernel 1: split fp32 weight into bf16 hi/lo parts.
// ---------------------------------------------------------------------------
__global__ void wsplit_kernel(const float* __restrict__ w,
                              bf16* __restrict__ whi, bf16* __restrict__ wlo, int n) {
  int i = blockIdx.x * blockDim.x + threadIdx.x;
  if (i < n) {
    float x = w[i];
    bf16 h = (bf16)x;
    whi[i] = h;
    wlo[i] = (bf16)(x - (float)h);
  }
}

// ---------------------------------------------------------------------------
// Kernel 2: per (b,t): n2 = sum_c e^2, dr = sum_c e[c,t]*e[c,t+1] (0 at t=T-1)
// ---------------------------------------------------------------------------
__global__ void embstat_kernel(const float* __restrict__ emb,
                               float* __restrict__ n2, float* __restrict__ dr) {
  int idx = blockIdx.x * blockDim.x + threadIdx.x;   // b*T + t
  int t = idx & (T_ - 1);
  const float* e = emb + (size_t)(idx - t) * C_ + t; // b*C*T + t
  float s2 = 0.f, sd = 0.f;
  const bool hasR = (t + 1) < T_;
  for (int c = 0; c < C_; ++c) {
    float v = e[(size_t)c * T_];
    s2 = fmaf(v, v, s2);
    float vr = hasR ? e[(size_t)c * T_ + 1] : 0.f;
    sd = fmaf(v, vr, sd);
  }
  n2[idx] = s2;
  dr[idx] = sd;
}

// ---------------------------------------------------------------------------
// Kernel 3: att[b, {l,c,r}, t] from n2/dr (sim_l[t] == sim_r[t-1]).
// ---------------------------------------------------------------------------
__global__ void att_kernel(const float* __restrict__ n2, const float* __restrict__ dr,
                           float* __restrict__ att) {
  int idx = blockIdx.x * blockDim.x + threadIdx.x;   // b*T + t
  int t = idx & (T_ - 1);
  int b = idx >> 12;
  float n2c = n2[idx];
  float nc  = fmaxf(sqrtf(n2c), EPS);
  float simc = n2c / (nc * nc);
  float siml = 0.f, simr = 0.f;
  if (t > 0) {
    float nl = fmaxf(sqrtf(n2[idx - 1]), EPS);
    siml = dr[idx - 1] / (nl * nc);
  }
  if (t + 1 < T_) {
    float nr = fmaxf(sqrtf(n2[idx + 1]), EPS);
    simr = dr[idx] / (nc * nr);
  }
  float* ab = att + (size_t)b * 3 * T_;
  ab[0 * T_ + t] = siml;
  ab[1 * T_ + t] = simc;
  ab[2 * T_ + t] = simr;
}

// ---------------------------------------------------------------------------
// CDNA5 async copy: global -> LDS, 16 bytes per lane, tracked by ASYNCcnt.
// LDS address = low 32 bits of the generic pointer (aperture: addr[31:0]).
// ---------------------------------------------------------------------------
__device__ __forceinline__ uint32_t lds_off_of(const void* p) {
  return (uint32_t)(uintptr_t)p;
}
__device__ __forceinline__ void async_ld_b128(uint32_t lds_off, const void* g) {
  asm volatile("global_load_async_to_lds_b128 %0, %1, off"
               :: "v"(lds_off), "v"((uint64_t)(uintptr_t)g)
               : "memory");
}
__device__ __forceinline__ void async_wait0() {
  asm volatile("s_wait_asynccnt 0x0" ::: "memory");
}

// ---------------------------------------------------------------------------
// Fragment assembly (ISA 7.12.2 16-bit layouts), LDS rows fragment-native.
// ---------------------------------------------------------------------------
__device__ __forceinline__ v16bf cat16(v8bf a, v8bf b) {
  return __builtin_shufflevector(a, b, 0,1,2,3,4,5,6,7,8,9,10,11,12,13,14,15);
}
__device__ __forceinline__ v16bf loadFragA(const bf16* row, int kA) {
  v8bf x = *(const v8bf*)(row + kA);
  v8bf y = *(const v8bf*)(row + kA + 16);
  return cat16(x, y);
}
__device__ __forceinline__ v16bf loadFragB(const bf16* row, int kB) {
  v8bf x = *(const v8bf*)(row + kB);
  v8bf y = *(const v8bf*)(row + kB + 16);
  return cat16(x, y);
}

// ---------------------------------------------------------------------------
// B staging: G tile (transposed [t][k]) = feature(im2col) * att, bf16 hi/lo.
// Thread -> 1 t-row (tl) x 8 k-cols (oct*8..+7); one v8bf store per array.
// Fast path: uniform SGPR base + 8 precomputed loop-invariant offsets.
// ---------------------------------------------------------------------------
struct Off8 { int v[8]; };

__device__ __forceinline__ void stageB_fast(const float* __restrict__ sg, Off8 off,
                                            float a, int oct, int tl,
                                            bf16* __restrict__ sBh,
                                            bf16* __restrict__ sBl) {
  v8bf h, l;
  #pragma unroll
  for (int e = 0; e < 8; ++e) {
    const float gv = sg[off.v[e]] * a;
    const bf16 hh = (bf16)gv;
    h[e] = hh;
    l[e] = (bf16)(gv - (float)hh);
  }
  *(v8bf*)(&sBh[tl * LDA + oct * 8]) = h;
  *(v8bf*)(&sBl[tl * LDA + oct * 8]) = l;
}

__device__ __forceinline__ void stageB_slow(const float* __restrict__ fb, int kb, int t0,
                                            float a, int oct, int tl,
                                            bf16* __restrict__ sBh,
                                            bf16* __restrict__ sBl) {
  v8bf h, l;
  #pragma unroll
  for (int e = 0; e < 8; ++e) {
    const int i = kb + oct * 8 + e;
    const int c = i / 3;
    const int k = i - 3 * c;
    const int tg = t0 + tl + k - 1;      // pad = 1
    const float f = ((unsigned)tg < (unsigned)T_) ? fb[(size_t)c * T_ + tg] : 0.f;
    const float gv = f * a;
    const bf16 hh = (bf16)gv;
    h[e] = hh;
    l[e] = (bf16)(gv - (float)hh);
  }
  *(v8bf*)(&sBh[tl * LDA + oct * 8]) = h;
  *(v8bf*)(&sBl[tl * LDA + oct * 8]) = l;
}

// ---------------------------------------------------------------------------
// Kernel 4: batched GEMM. Block = 256 threads = 8 waves (4 M x 2 N); each wave
// owns 64x32 = 4x2 subtiles of 16x16; 3 wmma per subtile per step (bf16x3).
// Loop unrolled by 3 (im2col pattern period: 32 ≡ 2 mod 3, 96K per group).
// Pipeline/step: frag loads -> barrier -> async A(next)+stage B(next)
//                -> 24x wmma -> s_wait_asynccnt -> barrier.
// ---------------------------------------------------------------------------
__global__ __launch_bounds__(256)
void gemm_kernel(const float* __restrict__ feat, const bf16* __restrict__ whi,
                 const bf16* __restrict__ wlo, const float* __restrict__ att,
                 float* __restrict__ out) {
  __shared__ __align__(16) bf16 sAh[BM * LDA];   // 20 KB
  __shared__ __align__(16) bf16 sAl[BM * LDA];   // 20 KB
  __shared__ __align__(16) bf16 sBh[BN * LDA];   //  5 KB
  __shared__ __align__(16) bf16 sBl[BN * LDA];   //  5 KB

  const int tid  = threadIdx.x;
  const int lane = tid & 31;
  const int wave = tid >> 5;       // 0..7
  const int wm   = wave >> 1;      // 0..3 : 64 rows (O) each
  const int wn   = wave & 1;       // 0..1 : 32 cols (T) each

  const int t0 = blockIdx.x * BN;
  const int o0 = blockIdx.y * BM;
  const int b  = blockIdx.z;
  const bool edge = (t0 == 0) || (t0 + BN >= T_);

  const float* fb  = feat + (size_t)b * C_ * T_;
  const float* fbt = fb + t0;                      // uniform base for fast staging
  const float* ab  = att + (size_t)b * 3 * T_;

  v8f acc[4][2];
  #pragma unroll
  for (int i = 0; i < 4; ++i)
    #pragma unroll
    for (int j = 0; j < 2; ++j)
      acc[i][j] = (v8f){0.f, 0.f, 0.f, 0.f, 0.f, 0.f, 0.f, 0.f};

  // A staging: thread -> one O row; 4 async b128 per hi/lo array per step.
  const uint32_t ah_off = lds_off_of(&sAh[tid * LDA]);
  const uint32_t al_off = lds_off_of(&sAl[tid * LDA]);
  const size_t   arow   = (size_t)(o0 + tid) * KK_;

  // B staging: thread -> (kk octet, one t row)
  const int oct = tid & 3;
  const int tl  = tid >> 2;        // 0..63

  // K-loop-invariant attention scalars (s = i>>9: torch repeat semantics).
  const float av0 = ab[0 * T_ + t0 + tl];
  const float av1 = ab[1 * T_ + t0 + tl];
  const float av2 = ab[2 * T_ + t0 + tl];

  // Loop-invariant im2col offsets: i = 96g + 32j + oct*8 + e,
  // c = 32g + ii/3, k = ii%3  ->  addr = fbt + g*32*T + off0[j][e]
  Off8 off0[3];
  #pragma unroll
  for (int j = 0; j < 3; ++j)
    #pragma unroll
    for (int e = 0; e < 8; ++e) {
      const int ii = 32 * j + oct * 8 + e;
      off0[j].v[e] = (ii / 3) * T_ + (ii % 3) - 1 + tl;
    }

  const int mrow = lane & 15;
  const int kA   = (lane < 16) ? 0 : 8;
  const int kB   = (lane < 16) ? 0 : 8;

  // ---- prologue: stage tile kb=0 ----
  #pragma unroll
  for (int q = 0; q < 4; ++q) {
    async_ld_b128(ah_off + 16 * q, whi + arow + 8 * q);
    async_ld_b128(al_off + 16 * q, wlo + arow + 8 * q);
  }
  if (!edge) stageB_fast(fbt, off0[0], av0, oct, tl, sBh, sBl);
  else       stageB_slow(fb, 0, t0, av0, oct, tl, sBh, sBl);
  async_wait0();
  __syncthreads();

  for (int g = 0; g < 16; ++g) {
    #pragma unroll
    for (int j = 0; j < 3; ++j) {
      // ---- fragments for current tile ----
      v16bf Ah[4], Al[4], Bh[2], Bl[2];
      #pragma unroll
      for (int mt = 0; mt < 4; ++mt) {
        const int r = (wm * 64 + mt * 16 + mrow) * LDA;
        Ah[mt] = loadFragA(&sAh[r], kA);
        Al[mt] = loadFragA(&sAl[r], kA);
      }
      #pragma unroll
      for (int nt = 0; nt < 2; ++nt) {
        const int r = (wn * 32 + nt * 16 + mrow) * LDA;
        Bh[nt] = loadFragB(&sBh[r], kB);
        Bl[nt] = loadFragB(&sBl[r], kB);
      }
      __syncthreads();   // all LDS reads of this tile done

      // ---- stage next tile (overlaps with wmma below) ----
      const int kb2 = (g * 3 + j + 1) * BK;
      if (kb2 < KK_) {
        #pragma unroll
        for (int q = 0; q < 4; ++q) {
          async_ld_b128(ah_off + 16 * q, whi + arow + kb2 + 8 * q);
          async_ld_b128(al_off + 16 * q, wlo + arow + kb2 + 8 * q);
        }
        const int s = kb2 >> 9;    // uniform per step
        const float a = (s == 0) ? av0 : (s == 1) ? av1 : av2;
        if (!edge) {
          const int   jn  = (j == 2) ? 0 : j + 1;           // compile-time
          const float* sg = fbt + (size_t)(j == 2 ? g + 1 : g) * (32 * (size_t)T_);
          stageB_fast(sg, off0[jn], a, oct, tl, sBh, sBl);
        } else {
          stageB_slow(fb, kb2, t0, a, oct, tl, sBh, sBl);
        }
      }

      // ---- 3-term bf16 split MAC: hi*hi + hi*lo + lo*hi (f32 accumulate) ----
      #pragma unroll
      for (int mt = 0; mt < 4; ++mt)
        #pragma unroll
        for (int nt = 0; nt < 2; ++nt) {
          acc[mt][nt] = __builtin_amdgcn_wmma_f32_16x16x32_bf16(
              false, Ah[mt], false, Bh[nt], (short)0, acc[mt][nt], false, false);
          acc[mt][nt] = __builtin_amdgcn_wmma_f32_16x16x32_bf16(
              false, Ah[mt], false, Bl[nt], (short)0, acc[mt][nt], false, false);
          acc[mt][nt] = __builtin_amdgcn_wmma_f32_16x16x32_bf16(
              false, Al[mt], false, Bh[nt], (short)0, acc[mt][nt], false, false);
        }

      if (kb2 < KK_) async_wait0();   // async A copy landed in LDS
      __syncthreads();                // staged tile visible to all waves
    }
  }

  // ---- epilogue: C/D layout -> lane = N col, VGPR r -> M = r + 8*(lane>=16) ----
  const int nlane = lane & 15;
  const int mhalf = (lane >> 4) * 8;
  float* ob = out + (size_t)b * O_ * T_;
  #pragma unroll
  for (int mt = 0; mt < 4; ++mt) {
    #pragma unroll
    for (int nt = 0; nt < 2; ++nt) {
      const int tcol  = t0 + wn * 32 + nt * 16 + nlane;
      const int obase = o0 + wm * 64 + mt * 16 + mhalf;
      #pragma unroll
      for (int r = 0; r < 8; ++r)
        ob[(size_t)(obase + r) * T_ + tcol] = acc[mt][nt][r];
    }
  }
}

// ---------------------------------------------------------------------------
// Launch: ws layout (bytes):
//   [0,        1572864) whi  bf16[512*1536]
//   [1572864,  3145728) wlo  bf16[512*1536]
//   [3145728,  3276800) n2   f32 [8*4096]
//   [3276800,  3407872) dr   f32 [8*4096]
//   [3407872,  3801088) att  f32 [8*3*4096]
// ---------------------------------------------------------------------------
extern "C" void kernel_launch(void* const* d_in, const int* in_sizes, int n_in,
                              void* d_out, int out_size, void* d_ws, size_t ws_size,
                              hipStream_t stream) {
  (void)in_sizes; (void)n_in; (void)out_size; (void)ws_size;
  const float* feature   = (const float*)d_in[0];
  const float* embedding = (const float*)d_in[1];
  const float* weight    = (const float*)d_in[2];
  float* out = (float*)d_out;

  char* ws = (char*)d_ws;
  bf16*  whi = (bf16*)(ws);
  bf16*  wlo = (bf16*)(ws + 1572864);
  float* n2  = (float*)(ws + 3145728);
  float* dr  = (float*)(ws + 3276800);
  float* att = (float*)(ws + 3407872);

  wsplit_kernel<<<dim3((O_ * KK_ + 255) / 256), dim3(256), 0, stream>>>(
      weight, whi, wlo, O_ * KK_);
  embstat_kernel<<<dim3((B_ * T_) / 256), dim3(256), 0, stream>>>(embedding, n2, dr);
  att_kernel<<<dim3((B_ * T_) / 256), dim3(256), 0, stream>>>(n2, dr, att);
  gemm_kernel<<<dim3(T_ / BN, O_ / BM, B_), dim3(256), 0, stream>>>(
      feature, whi, wlo, att, out);
}
